// ConCH_18717467476370
// MI455X (gfx1250) — compile-verified, matrix operand order checked by reading, not run on hardware
//
#include <hip/hip_runtime.h>
#include <hip/hip_bf16.h>
#include <math.h>
#include <stdint.h>

#define NN   10000
#define FIN  3000
#define FINP 3008
#define HH1  500
#define H1P  512
#define HH2  64
#define EE   160000
#define ZNPAD 10048   // 157*64, pad for rec_adj B-tile coverage

typedef __attribute__((ext_vector_type(16))) __bf16 v16bf;
typedef __attribute__((ext_vector_type(8)))  float  v8f;
typedef __attribute__((ext_vector_type(2)))  float  v2f;

union FragBF { uint4 u[2]; v16bf v; };

__device__ __forceinline__ __bf16 f2bf(float f) {
  unsigned u = __builtin_bit_cast(unsigned, f);
  unsigned r = (u + 0x7FFFu + ((u >> 16) & 1u)) >> 16;
  unsigned short s = (unsigned short)r;
  return __builtin_bit_cast(__bf16, s);
}

__device__ __forceinline__ void ldfrag(FragBF& f, const __bf16* p) {
  f.u[0] = *(const uint4*)(p);
  f.u[1] = *(const uint4*)(p + 16);
}

enum { EPI_NONE = 0, EPI_SIGMOID = 1, EPI_SOFTPLUS = 2, EPI_EXP = 3, EPI_BN_RELU = 4 };

// ---------------------------------------------------------------------------
// bf16 WMMA GEMM: C[M x Nc] = A[M x K](bf16,lda) * Bt[* x K](bf16,ldbt)^T
// Block = 256 thr (8 waves); wave tile = 16(M) x 64(N); block tile = 128 x 64.
// Requires K % 64 == 0 (all call sites pad K). Software-pipelined ping-pong
// fragment sets so WMMA only waits on its own fragment loads.
// ---------------------------------------------------------------------------
template <int EPI, bool OUT_BF16, bool NT>
__global__ __launch_bounds__(256) void gemm_bf16_ker(
    const __bf16* __restrict__ A, const __bf16* __restrict__ Bt, void* __restrict__ Cout,
    const float* __restrict__ bias, const float* __restrict__ bng, const float* __restrict__ bnb,
    const float* __restrict__ bnm, const float* __restrict__ bnv,
    int M, int Nc, int K, int lda, int ldbt, int ldc)
{
  const int lane = threadIdx.x & 31;
  const int wv   = threadIdx.x >> 5;
  const int hl   = lane >> 4;          // half-wave select
  const int l16  = lane & 15;
  const int rowTile = blockIdx.y * 128 + wv * 16;
  int arow = rowTile + l16; if (arow > M - 1) arow = M - 1;   // clamp, mask at store
  const int colTile = blockIdx.x * 64;
  const int koff = hl * 8;             // lanes 16..31 start at K+8 (ISA A/B layout)

  const __bf16* ab = A + (size_t)arow * lda + koff;
  const __bf16* bb[4];
  #pragma unroll
  for (int j = 0; j < 4; ++j) bb[j] = Bt + (size_t)(colTile + j * 16 + l16) * ldbt + koff;

  const v8f zero = {0.f,0.f,0.f,0.f,0.f,0.f,0.f,0.f};
  v8f acc[4] = {zero, zero, zero, zero};

  FragBF a0, a1, b0[4], b1[4];
  ldfrag(a0, ab);
  #pragma unroll
  for (int j = 0; j < 4; ++j) ldfrag(b0[j], bb[j]);

  int k = 0;
  for (; k + 64 < K; k += 64) {
    // prefetch set1 (k+32) before consuming set0 (k)
    ldfrag(a1, ab + k + 32);
    #pragma unroll
    for (int j = 0; j < 4; ++j) ldfrag(b1[j], bb[j] + k + 32);
    #pragma unroll
    for (int j = 0; j < 4; ++j)
      acc[j] = __builtin_amdgcn_wmma_f32_16x16x32_bf16(false, a0.v, false, b0[j].v, (short)0, acc[j], false, false);
    // prefetch set0 (k+64) before consuming set1 (k+32)
    ldfrag(a0, ab + k + 64);
    #pragma unroll
    for (int j = 0; j < 4; ++j) ldfrag(b0[j], bb[j] + k + 64);
    #pragma unroll
    for (int j = 0; j < 4; ++j)
      acc[j] = __builtin_amdgcn_wmma_f32_16x16x32_bf16(false, a1.v, false, b1[j].v, (short)0, acc[j], false, false);
  }
  // tail: k == K - 64
  ldfrag(a1, ab + k + 32);
  #pragma unroll
  for (int j = 0; j < 4; ++j) ldfrag(b1[j], bb[j] + k + 32);
  #pragma unroll
  for (int j = 0; j < 4; ++j)
    acc[j] = __builtin_amdgcn_wmma_f32_16x16x32_bf16(false, a0.v, false, b0[j].v, (short)0, acc[j], false, false);
  #pragma unroll
  for (int j = 0; j < 4; ++j)
    acc[j] = __builtin_amdgcn_wmma_f32_16x16x32_bf16(false, a1.v, false, b1[j].v, (short)0, acc[j], false, false);

  const int mBase = rowTile + hl * 8;  // C layout: VGPR v -> row v (+8 for high half)
  #pragma unroll
  for (int b = 0; b < 4; ++b) {
    const int n = colTile + b * 16 + l16;
    const bool nOK = (n < Nc);
    float bi = (nOK && bias) ? bias[n] : 0.f;
    float sc = 1.f, sh = 0.f;
    if (EPI == EPI_BN_RELU && nOK) {
      sc = bng[n] * rsqrtf(bnv[n] + 1e-5f);
      sh = bnb[n] - bnm[n] * sc;
    }
    #pragma unroll
    for (int v = 0; v < 8; ++v) {
      const int m = mBase + v;
      if (m >= M) continue;
      float x = acc[b][v] + bi;
      if (EPI == EPI_SIGMOID)  x = 1.f / (1.f + expf(-x));
      if (EPI == EPI_SOFTPLUS) { float sp = (x > 20.f) ? x : log1pf(expf(x)); x = fminf(fmaxf(sp, 1e-4f), 1e4f); }
      if (EPI == EPI_EXP)      x = fminf(fmaxf(expf(x), 1e-5f), 1e6f);
      if (EPI == EPI_BN_RELU)  x = fmaxf(x * sc + sh, 0.f);
      if (OUT_BF16) {
        ((__bf16*)Cout)[(size_t)m * ldc + n] = f2bf(nOK ? x : 0.f);  // zero the pad cols
      } else if (nOK) {
        float* cp = ((float*)Cout) + (size_t)m * ldc + n;
        if (NT) __builtin_nontemporal_store(x, cp);
        else    *cp = x;
      }
    }
  }
}

// ---------------------------------------------------------------------------
// f32 WMMA GEMM (16x16x4) for graph_neigh @ emb1 : A fp32 direct from HBM.
// Requires K % 8 == 0. Ping-pong pipelined like the bf16 kernel.
// ---------------------------------------------------------------------------
__global__ __launch_bounds__(256) void gemm_f32x4_ker(
    const float* __restrict__ A, const float* __restrict__ Bt, float* __restrict__ C,
    int M, int Nc, int K, int lda, int ldbt, int ldc)
{
  const int lane = threadIdx.x & 31;
  const int wv   = threadIdx.x >> 5;
  const int hl   = lane >> 4;
  const int l16  = lane & 15;
  const int rowTile = blockIdx.y * 128 + wv * 16;
  int arow = rowTile + l16; if (arow > M - 1) arow = M - 1;
  const int colTile = blockIdx.x * 64;
  const int koff = hl * 2;             // 32-bit A 16x4: lanes 16..31 hold K=2,3

  const float* ab = A + (size_t)arow * lda + koff;
  const float* bb[4];
  #pragma unroll
  for (int b = 0; b < 4; ++b) bb[b] = Bt + (size_t)(colTile + b * 16 + l16) * ldbt + koff;

  const v8f zero = {0.f,0.f,0.f,0.f,0.f,0.f,0.f,0.f};
  v8f acc[4] = {zero, zero, zero, zero};

  v2f a0, a1, b0[4], b1[4];
  a0 = *(const v2f*)(ab);
  #pragma unroll
  for (int j = 0; j < 4; ++j) b0[j] = *(const v2f*)(bb[j]);

  int k = 0;
  for (; k + 8 < K; k += 8) {
    a1 = *(const v2f*)(ab + k + 4);
    #pragma unroll
    for (int j = 0; j < 4; ++j) b1[j] = *(const v2f*)(bb[j] + k + 4);
    #pragma unroll
    for (int j = 0; j < 4; ++j)
      acc[j] = __builtin_amdgcn_wmma_f32_16x16x4_f32(false, a0, false, b0[j], (short)0, acc[j], false, false);
    a0 = *(const v2f*)(ab + k + 8);
    #pragma unroll
    for (int j = 0; j < 4; ++j) b0[j] = *(const v2f*)(bb[j] + k + 8);
    #pragma unroll
    for (int j = 0; j < 4; ++j)
      acc[j] = __builtin_amdgcn_wmma_f32_16x16x4_f32(false, a1, false, b1[j], (short)0, acc[j], false, false);
  }
  // tail: k == K - 8
  a1 = *(const v2f*)(ab + k + 4);
  #pragma unroll
  for (int j = 0; j < 4; ++j) b1[j] = *(const v2f*)(bb[j] + k + 4);
  #pragma unroll
  for (int j = 0; j < 4; ++j)
    acc[j] = __builtin_amdgcn_wmma_f32_16x16x4_f32(false, a0, false, b0[j], (short)0, acc[j], false, false);
  #pragma unroll
  for (int j = 0; j < 4; ++j)
    acc[j] = __builtin_amdgcn_wmma_f32_16x16x4_f32(false, a1, false, b1[j], (short)0, acc[j], false, false);

  const int mBase = rowTile + hl * 8;
  #pragma unroll
  for (int b = 0; b < 4; ++b) {
    const int n = colTile + b * 16 + l16;
    if (n >= Nc) continue;
    #pragma unroll
    for (int v = 0; v < 8; ++v) {
      const int m = mBase + v;
      if (m < M) C[(size_t)m * ldc + n] = acc[b][v];
    }
  }
}

// --------------------------- support kernels --------------------------------
__global__ void fill_f32_ker(float* p, long long n) {
  long long i = (long long)blockIdx.x * blockDim.x + threadIdx.x;
  for (; i < n; i += (long long)gridDim.x * blockDim.x) p[i] = 0.f;
}
__global__ void fill_bf16_ker(__bf16* p, long long n) {
  long long i = (long long)blockIdx.x * blockDim.x + threadIdx.x;
  for (; i < n; i += (long long)gridDim.x * blockDim.x) p[i] = f2bf(0.f);
}

// src [Mr x K] f32 row-major -> dst [Mr x Kpad] bf16, zero pad, optional relu
__global__ void cast_pad_ker(const float* __restrict__ src, __bf16* __restrict__ dst,
                             int Mr, int K, int Kpad, int relu) {
  long long idx = (long long)blockIdx.x * blockDim.x + threadIdx.x;
  long long total = (long long)Mr * Kpad;
  if (idx >= total) return;
  int m = (int)(idx / Kpad), k = (int)(idx % Kpad);
  float v = (k < K) ? src[(size_t)m * K + k] : 0.f;
  if (relu) v = fmaxf(v, 0.f);
  dst[idx] = f2bf(v);
}

// W [K x Nw] f32 -> Wt [Npad x Kpad] bf16 (transposed, zero-padded)
__global__ void transpose_cast_ker(const float* __restrict__ W, __bf16* __restrict__ Wt,
                                   int K, int Nw, int Kpad, int Npad) {
  long long idx = (long long)blockIdx.x * blockDim.x + threadIdx.x;
  long long total = (long long)Npad * Kpad;
  if (idx >= total) return;
  int n = (int)(idx / Kpad), k = (int)(idx % Kpad);
  float v = (n < Nw && k < K) ? W[(size_t)k * Nw + n] : 0.f;
  Wt[idx] = f2bf(v);
}

// out[rows[e]][f] += vals[e] * sup[cols[e]][f]  (H compile-time: no runtime div)
template <int H>
__global__ void aggregate_ker(const int* __restrict__ rows, const int* __restrict__ cols,
                              const float* __restrict__ vals, const float* __restrict__ sup,
                              float* __restrict__ out, int E) {
  long long idx = (long long)blockIdx.x * blockDim.x + threadIdx.x;
  if (idx >= (long long)E * H) return;
  int e = (int)(idx / H), f = (int)(idx % H);
  atomicAdd(&out[(size_t)rows[e] * H + f], vals[e] * sup[(size_t)cols[e] * H + f]);
}

// zn = l2norm(z1) rows of 64, stored bf16
__global__ void l2norm_bf16_ker(const float* __restrict__ z, __bf16* __restrict__ zn, int Nn) {
  int r = blockIdx.x * blockDim.x + threadIdx.x;
  if (r >= Nn) return;
  const float* p = z + (size_t)r * 64;
  float s = 0.f;
  for (int d = 0; d < 64; ++d) s += p[d] * p[d];
  float den = fmaxf(sqrtf(s), 1e-12f);
  for (int d = 0; d < 64; ++d) zn[(size_t)r * 64 + d] = f2bf(p[d] / den);
}

// embT[d][n] = relu(z[n][d]),  embT is [64 x Nn] f32 (= Bt for f32 GEMM)
__global__ void transpose_relu_ker(const float* __restrict__ z, float* __restrict__ embT, int Nn) {
  long long idx = (long long)blockIdx.x * blockDim.x + threadIdx.x;
  if (idx >= (long long)64 * Nn) return;
  int d = (int)(idx / Nn), n = (int)(idx % Nn);
  embT[idx] = fmaxf(z[(size_t)n * 64 + d], 0.f);
}

// wave-per-row rowsum of graph_neigh [Nn x Nn]
__global__ __launch_bounds__(256) void rowsum_ker(const float* __restrict__ g, float* __restrict__ rs, int Nn) {
  int gtid = blockIdx.x * 256 + threadIdx.x;
  int row = gtid >> 5, lane = threadIdx.x & 31;
  if (row >= Nn) return;
  const float* p = g + (size_t)row * Nn;
  float s = 0.f;
  for (int j = lane; j < Nn; j += 32) s += p[j];
  for (int off = 16; off > 0; off >>= 1) s += __shfl_down(s, off, 32);
  if (lane == 0) rs[row] = s;
}

// g2 = sigmoid(l2norm(vsum / rowsum))
__global__ void g2_ker(const float* __restrict__ vsum, const float* __restrict__ rs,
                       float* __restrict__ g2, int Nn) {
  int r = blockIdx.x * blockDim.x + threadIdx.x;
  if (r >= Nn) return;
  float inv = 1.f / rs[r];
  const float* p = vsum + (size_t)r * 64;
  float s = 0.f;
  for (int d = 0; d < 64; ++d) { float x = p[d] * inv; s += x * x; }
  float den = fmaxf(sqrtf(s), 1e-12f);
  for (int d = 0; d < 64; ++d)
    g2[(size_t)r * 64 + d] = 1.f / (1.f + expf(-(p[d] * inv / den)));
}

// bilinear discriminator: sc = relu(z)^T (discW @ g2) + b ; one block per node
__global__ void scores_ker(const float* __restrict__ z1, const float* __restrict__ z3,
                           const float* __restrict__ g2, const float* __restrict__ dW,
                           const float* __restrict__ db, float* __restrict__ ret) {
  __shared__ float gs[64], p1[64], p3[64];
  int n = blockIdx.x, d = threadIdx.x;
  gs[d] = g2[(size_t)n * 64 + d];
  __syncthreads();
  float t = 0.f;
  #pragma unroll 8
  for (int e = 0; e < 64; ++e) t += dW[d * 64 + e] * gs[e];
  p1[d] = fmaxf(z1[(size_t)n * 64 + d], 0.f) * t;
  p3[d] = fmaxf(z3[(size_t)n * 64 + d], 0.f) * t;
  __syncthreads();
  for (int s = 32; s > 0; s >>= 1) {
    if (d < s) { p1[d] += p1[d + s]; p3[d] += p3[d + s]; }
    __syncthreads();
  }
  if (d == 0) {
    ret[(size_t)n * 2 + 0] = p1[0] + db[0];
    ret[(size_t)n * 2 + 1] = p3[0] + db[0];
  }
}

// ---------------------------------------------------------------------------
extern "C" void kernel_launch(void* const* d_in, const int* in_sizes, int n_in,
                              void* d_out, int out_size, void* d_ws, size_t ws_size,
                              hipStream_t stream) {
  (void)in_sizes; (void)n_in; (void)out_size; (void)ws_size;
  const float* feat   = (const float*)d_in[0];
  const float* feat_a = (const float*)d_in[1];
  const float* feat_b = (const float*)d_in[2];
  const int*   arows  = (const int*)d_in[3];
  const int*   acols  = (const int*)d_in[4];
  const float* avals  = (const float*)d_in[5];
  const float* graph  = (const float*)d_in[6];
  const float* W1     = (const float*)d_in[7];
  const float* W2     = (const float*)d_in[8];
  const float* Wd     = (const float*)d_in[9];
  const float* bd     = (const float*)d_in[10];
  const float* bng    = (const float*)d_in[11];
  const float* bnb    = (const float*)d_in[12];
  const float* bnm    = (const float*)d_in[13];
  const float* bnv    = (const float*)d_in[14];
  const float* Wpi    = (const float*)d_in[15];
  const float* bpi    = (const float*)d_in[16];
  const float* Wdisp  = (const float*)d_in[17];
  const float* bdisp  = (const float*)d_in[18];
  const float* Wmean  = (const float*)d_in[19];
  const float* bmean  = (const float*)d_in[20];
  const float* discW  = (const float*)d_in[21];
  const float* discb  = (const float*)d_in[22];

  float* out = (float*)d_out;
  float* z1o = out;                       // [N,64]
  float* z2o = out + 640000;
  float* z3o = out + 1280000;
  float* pio = out + 1920000;             // [N,3000]
  float* dpo = out + 31920000;
  float* mno = out + 61920000;
  float* rco = out + 91920000;            // [N,N]
  float* rto = out + 191920000;           // [N,2]

  char* wsp = (char*)d_ws; size_t off = 0;
  auto carve = [&](size_t bytes) -> char* {
    char* p = wsp + off; off = (off + bytes + 255) & ~(size_t)255; return p;
  };
  __bf16* featbf = (__bf16*)carve((size_t)NN * FINP * 2);
  __bf16* W1t    = (__bf16*)carve((size_t)H1P * FINP * 2);
  __bf16* W2t    = (__bf16*)carve((size_t)64 * H1P * 2);
  __bf16* Wdt    = (__bf16*)carve((size_t)H1P * 64 * 2);
  __bf16* Wpit   = (__bf16*)carve((size_t)FINP * H1P * 2);
  __bf16* Wdpt   = (__bf16*)carve((size_t)FINP * H1P * 2);
  __bf16* Wmnt   = (__bf16*)carve((size_t)FINP * H1P * 2);
  float*  sup1   = (float*)carve((size_t)NN * HH1 * 4);
  float*  agg1   = (float*)carve((size_t)NN * HH1 * 4);
  __bf16* hbuf   = (__bf16*)carve((size_t)NN * H1P * 2);
  float*  sup2   = (float*)carve((size_t)NN * 64 * 4);
  __bf16* z1bf   = (__bf16*)carve((size_t)NN * 64 * 2);
  __bf16* xdbf   = (__bf16*)carve((size_t)NN * H1P * 2);
  __bf16* znbf   = (__bf16*)carve((size_t)ZNPAD * 64 * 2);
  float*  embT   = (float*)carve((size_t)64 * NN * 4);
  float*  vsum   = (float*)carve((size_t)NN * 64 * 4);
  float*  rsum   = (float*)carve((size_t)NN * 4);
  float*  g2b    = (float*)carve((size_t)NN * 64 * 4);

  auto blks = [](long long n) { return (unsigned)((n + 255) / 256); };
  const dim3 TB(256);
  const unsigned MGRID = (NN + 127) / 128;   // 79

  // ---- weight prep (transposed bf16, padded) ----
  transpose_cast_ker<<<blks((long long)H1P * FINP), TB, 0, stream>>>(W1, W1t, FIN, HH1, FINP, H1P);
  transpose_cast_ker<<<blks((long long)64 * H1P),  TB, 0, stream>>>(W2, W2t, HH1, 64, H1P, 64);
  transpose_cast_ker<<<blks((long long)H1P * 64),  TB, 0, stream>>>(Wd, Wdt, 64, HH1, 64, H1P);
  transpose_cast_ker<<<blks((long long)FINP * H1P), TB, 0, stream>>>(Wpi,   Wpit, HH1, FIN, H1P, FINP);
  transpose_cast_ker<<<blks((long long)FINP * H1P), TB, 0, stream>>>(Wdisp, Wdpt, HH1, FIN, H1P, FINP);
  transpose_cast_ker<<<blks((long long)FINP * H1P), TB, 0, stream>>>(Wmean, Wmnt, HH1, FIN, H1P, FINP);

  // ---- 3x encode: z = Agg(relu(Agg(X@W1)) @ W2) ----
  const float* Xs[3] = {feat, feat_a, feat_b};
  float* Zs[3] = {z1o, z2o, z3o};
  for (int i = 0; i < 3; ++i) {
    cast_pad_ker<<<blks((long long)NN * FINP), TB, 0, stream>>>(Xs[i], featbf, NN, FIN, FINP, 0);
    gemm_bf16_ker<EPI_NONE, false, false><<<dim3(H1P / 64, MGRID), TB, 0, stream>>>(
        featbf, W1t, sup1, nullptr, nullptr, nullptr, nullptr, nullptr,
        NN, HH1, FINP, FINP, FINP, HH1);
    fill_f32_ker<<<2048, TB, 0, stream>>>(agg1, (long long)NN * HH1);
    aggregate_ker<HH1><<<blks((long long)EE * HH1), TB, 0, stream>>>(arows, acols, avals, sup1, agg1, EE);
    cast_pad_ker<<<blks((long long)NN * H1P), TB, 0, stream>>>(agg1, hbuf, NN, HH1, H1P, 1);
    gemm_bf16_ker<EPI_NONE, false, false><<<dim3(1, MGRID), TB, 0, stream>>>(
        hbuf, W2t, sup2, nullptr, nullptr, nullptr, nullptr, nullptr,
        NN, 64, H1P, H1P, H1P, 64);
    fill_f32_ker<<<2048, TB, 0, stream>>>(Zs[i], (long long)NN * 64);
    aggregate_ker<64><<<blks((long long)EE * 64), TB, 0, stream>>>(arows, acols, avals, sup2, Zs[i], EE);
  }

  // ---- ZINB decoder: xd = relu(BN(z1@Wd + bd)) -> pi/disp/mean heads ----
  cast_pad_ker<<<blks((long long)NN * 64), TB, 0, stream>>>(z1o, z1bf, NN, 64, 64, 0);
  gemm_bf16_ker<EPI_BN_RELU, true, false><<<dim3(H1P / 64, MGRID), TB, 0, stream>>>(
      z1bf, Wdt, xdbf, bd, bng, bnb, bnm, bnv, NN, HH1, 64, 64, 64, H1P);
  gemm_bf16_ker<EPI_SIGMOID, false, true><<<dim3(FINP / 64, MGRID), TB, 0, stream>>>(
      xdbf, Wpit, pio, bpi, nullptr, nullptr, nullptr, nullptr, NN, FIN, H1P, H1P, H1P, FIN);
  gemm_bf16_ker<EPI_SOFTPLUS, false, true><<<dim3(FINP / 64, MGRID), TB, 0, stream>>>(
      xdbf, Wdpt, dpo, bdisp, nullptr, nullptr, nullptr, nullptr, NN, FIN, H1P, H1P, H1P, FIN);
  gemm_bf16_ker<EPI_EXP, false, true><<<dim3(FINP / 64, MGRID), TB, 0, stream>>>(
      xdbf, Wmnt, mno, bmean, nullptr, nullptr, nullptr, nullptr, NN, FIN, H1P, H1P, H1P, FIN);

  // ---- rec_adj = sigmoid(zn @ zn^T), NT stores (400 MB write-once) ----
  fill_bf16_ker<<<2048, TB, 0, stream>>>(znbf, (long long)ZNPAD * 64);
  l2norm_bf16_ker<<<blks(NN), TB, 0, stream>>>(z1o, znbf, NN);
  gemm_bf16_ker<EPI_SIGMOID, false, true><<<dim3(ZNPAD / 64, MGRID), TB, 0, stream>>>(
      znbf, znbf, rco, nullptr, nullptr, nullptr, nullptr, nullptr,
      NN, NN, 64, 64, 64, NN);

  // ---- readout: vsum = graph_neigh @ relu(z1)  (f32 WMMA, A direct from HBM) ----
  transpose_relu_ker<<<blks((long long)64 * NN), TB, 0, stream>>>(z1o, embT, NN);
  gemm_f32x4_ker<<<dim3(1, MGRID), TB, 0, stream>>>(graph, embT, vsum, NN, 64, NN, NN, NN, 64);
  rowsum_ker<<<blks((long long)NN * 32), TB, 0, stream>>>(graph, rsum, NN);
  g2_ker<<<blks(NN), TB, 0, stream>>>(vsum, rsum, g2b, NN);

  // ---- bilinear discriminator scores -> ret1 [N,2] ----
  scores_ker<<<NN, 64, 0, stream>>>(z1o, z3o, g2b, discW, discb, rto);
}
